// MultiHeadLSRAttention_63831803953614
// MI455X (gfx1250) — matrix operand order repaired and checked by reading, not compiled, not run on hardware
//
#include <hip/hip_runtime.h>

typedef __attribute__((ext_vector_type(16))) _Float16 v16h;
typedef __attribute__((ext_vector_type(8)))  _Float16 v8h;
typedef __attribute__((ext_vector_type(8)))  float    v8f;

#define BATCH 2
#define HEADS 16
#define SEQ   2048
#define DIM   64
#define RANK  32
#define BH    (BATCH * HEADS)

#define LOG2E 1.44269504088896340736f
#define NEG_BIG (-1.0e30f)

// ---------------------------------------------------------------------------
// 16-lane butterfly max using DPP (stays inside each 16-lane row; no LDS).
// quad_perm xor1 (0xB1), quad_perm xor2 (0x4E), row_ror:4 (0x124), row_ror:8 (0x128)
// ---------------------------------------------------------------------------
__device__ __forceinline__ float rowmax16(float x) {
  int i;
  i = __builtin_amdgcn_update_dpp(0, __builtin_bit_cast(int, x), 0xB1, 0xF, 0xF, true);
  x = fmaxf(x, __builtin_bit_cast(float, i));
  i = __builtin_amdgcn_update_dpp(0, __builtin_bit_cast(int, x), 0x4E, 0xF, 0xF, true);
  x = fmaxf(x, __builtin_bit_cast(float, i));
  i = __builtin_amdgcn_update_dpp(0, __builtin_bit_cast(int, x), 0x124, 0xF, 0xF, true);
  x = fmaxf(x, __builtin_bit_cast(float, i));
  i = __builtin_amdgcn_update_dpp(0, __builtin_bit_cast(int, x), 0x128, 0xF, 0xF, true);
  x = fmaxf(x, __builtin_bit_cast(float, i));
  return x;
}

__device__ __forceinline__ v16h make16(const v8h a, const v8h b) {
  v16h r;
#pragma unroll
  for (int i = 0; i < 8; ++i) { r[i] = a[i]; r[8 + i] = b[i]; }
  return r;
}

// ---------------------------------------------------------------------------
// Kernel 1: low-rank projections.  One wave per (b,h,t) row; lane r in [0,32)
// computes q_lr[row,r] and k_lr[row,r].  q_lr absorbs kron * 1/sqrt(R) * LOG2E
// so attention scores come out of WMMA already in the log2 domain.
// ---------------------------------------------------------------------------
__global__ __launch_bounds__(256)
void lsr_proj_kernel(const float* __restrict__ q, const float* __restrict__ k,
                     const float* __restrict__ Wq, const float* __restrict__ Wk,
                     const float* __restrict__ core1, const float* __restrict__ core2,
                     _Float16* __restrict__ qlr, _Float16* __restrict__ klr) {
  const int wave = threadIdx.x >> 5;
  const int lane = threadIdx.x & 31;
  const long row = (long)blockIdx.x * 8 + wave;          // [0, BH*SEQ)
  const int h = (int)((row / SEQ) % HEADS);

  const float* __restrict__ qrow = q + row * DIM;
  const float* __restrict__ krow = k + row * DIM;
  const float* __restrict__ wq = Wq + (long)h * DIM * RANK + lane;
  const float* __restrict__ wk = Wk + (long)h * DIM * RANK + lane;

  float aq = 0.0f, ak = 0.0f;
#pragma unroll 8
  for (int d = 0; d < DIM; ++d) {
    const float qv = qrow[d];
    const float kv = krow[d];
    aq = fmaf(qv, wq[d * RANK], aq);
    ak = fmaf(kv, wk[d * RANK], ak);
  }
  const float kron = core1[h * 4 + (lane >> 3)] * core2[h * 8 + (lane & 7)];
  const float scale = 0.17677669529663687f * LOG2E;      // 1/sqrt(32) * log2(e)
  qlr[row * RANK + lane] = (_Float16)(aq * kron * scale);
  klr[row * RANK + lane] = (_Float16)ak;
}

// ---------------------------------------------------------------------------
// Kernel 2: V -> V^T (f16) via 64x64 LDS tile transpose.
// ---------------------------------------------------------------------------
__global__ __launch_bounds__(256)
void lsr_vtrans_kernel(const float* __restrict__ v, _Float16* __restrict__ vt) {
  __shared__ float tile[64][65];
  const int bh = blockIdx.y;
  const int t0 = blockIdx.x * 64;
  const float* __restrict__ vb = v + (long)bh * SEQ * DIM;
  _Float16* __restrict__ vtb = vt + (long)bh * DIM * SEQ;

  const int tx = threadIdx.x & 63;
  const int ty = threadIdx.x >> 6;
#pragma unroll
  for (int i = 0; i < 64; i += 4)
    tile[ty + i][tx] = vb[(long)(t0 + ty + i) * DIM + tx];
  __syncthreads();
#pragma unroll
  for (int i = 0; i < 64; i += 4)
    vtb[(long)(ty + i) * SEQ + (t0 + tx)] = (_Float16)tile[tx][ty + i];
}

// ---------------------------------------------------------------------------
// One 64-key chunk of the flash-attention loop for one wave (16 query rows).
//   4x WMMA  : S tiles (A = q_lr 16x32, B = k_lr^T 32x16)
//   softmax  : log2-domain, DPP row-max, no cross-lane sum
//   bounce   : P f16 C-layout -> A-layout via per-wave LDS (s_wait_dscnt);
//              f16 staging so the A-tile reload is 4x ds_load_b128, no cvt
//   8x WMMA  : O += P @ V (B = V^T chunks)
//   2x WMMA  : L += P @ ones   (row-sum broadcast for free in C layout)
// ---------------------------------------------------------------------------
template <bool MASKED>
__device__ __forceinline__ void lsr_chunk(
    int nc, int m0, int lh, int ll,
    const _Float16* __restrict__ klrb, const _Float16* __restrict__ vtb,
    const v16h aQ, const v16h bOnes, _Float16* __restrict__ pbase,
    v8f (&O)[4], v8f& Lacc, float (&mrow)[8]) {
  // ---- S = q_lr @ k_lr^T for 64 keys (4 tiles of 16).
  v8f S[4];
#pragma unroll
  for (int t = 0; t < 4; ++t) {
    const _Float16* kr = klrb + (long)(nc + 16 * t + ll) * RANK + lh * 16;
    const v16h bK = make16(*(const v8h*)kr, *(const v8h*)(kr + 8));
    const v8f zero = {};
    S[t] = __builtin_amdgcn_wmma_f32_16x16x32_f16(false, aQ, false, bK,
                                                  (short)0, zero, false, false);
  }

  if (MASKED) {
#pragma unroll
    for (int t = 0; t < 4; ++t) {
      const int n = nc + 16 * t + ll;
#pragma unroll
      for (int j = 0; j < 8; ++j) {
        const int m = m0 + j + 8 * lh;
        S[t][j] = (n <= m) ? S[t][j] : NEG_BIG;
      }
    }
  }

  // ---- online softmax in log2 domain (scores already scaled by log2e).
#pragma unroll
  for (int j = 0; j < 8; ++j) {
    float mx = fmaxf(fmaxf(S[0][j], S[1][j]), fmaxf(S[2][j], S[3][j]));
    mx = rowmax16(mx);
    const float mnew  = fmaxf(mrow[j], mx);
    const float alpha = __builtin_amdgcn_exp2f(mrow[j] - mnew);
    mrow[j] = mnew;
    O[0][j] *= alpha; O[1][j] *= alpha; O[2][j] *= alpha; O[3][j] *= alpha;
    Lacc[j] *= alpha;
    S[0][j] = __builtin_amdgcn_exp2f(S[0][j] - mnew);
    S[1][j] = __builtin_amdgcn_exp2f(S[1][j] - mnew);
    S[2][j] = __builtin_amdgcn_exp2f(S[2][j] - mnew);
    S[3][j] = __builtin_amdgcn_exp2f(S[3][j] - mnew);
  }

  // ---- P bounce: C-layout -> A-layout through per-wave LDS [16][64] f16.
#pragma unroll
  for (int t = 0; t < 4; ++t)
#pragma unroll
    for (int j = 0; j < 8; ++j)
      pbase[(j + 8 * lh) * 64 + 16 * t + ll] = (_Float16)S[t][j];
  asm volatile("s_wait_dscnt 0" ::: "memory");

  v16h aP[2];
#pragma unroll
  for (int s = 0; s < 2; ++s) {
    const _Float16* b = pbase + ll * 64 + 32 * s;
    aP[s] = make16(*(const v8h*)(b + lh * 16), *(const v8h*)(b + 8 + lh * 16));
  }

  // ---- O += P @ V ; L += P @ ones.
#pragma unroll
  for (int s = 0; s < 2; ++s) {
#pragma unroll
    for (int c = 0; c < 4; ++c) {
      const _Float16* vb = vtb + (long)(c * 16 + ll) * SEQ + (nc + 32 * s) + lh * 16;
      const v16h bV = make16(*(const v8h*)vb, *(const v8h*)(vb + 8));
      O[c] = __builtin_amdgcn_wmma_f32_16x16x32_f16(false, aP[s], false, bV,
                                                    (short)0, O[c], false, false);
    }
    Lacc = __builtin_amdgcn_wmma_f32_16x16x32_f16(false, aP[s], false, bOnes,
                                                  (short)0, Lacc, false, false);
  }
}

// ---------------------------------------------------------------------------
// Kernel 3: causal flash attention on rank-32 scores.
// Block = 128 threads = 4 waves; wave w owns query rows [m0, m0+15].
// Full (unmasked) 64-key chunks, then exactly one masked diagonal chunk.
// No __syncthreads in the causal loop (each wave bounces P through its own LDS slice).
// ---------------------------------------------------------------------------
__global__ __launch_bounds__(128)
void lsr_attn_kernel(const _Float16* __restrict__ qlr,
                     const _Float16* __restrict__ klr,
                     const _Float16* __restrict__ vt,
                     float* __restrict__ out) {
  __shared__ _Float16 plds[4][16][64];   // 8 KB

  const int bh   = blockIdx.y;
  const int wave = threadIdx.x >> 5;
  const int lane = threadIdx.x & 31;
  const int lh   = lane >> 4;
  const int ll   = lane & 15;
  const int m0   = blockIdx.x * 64 + wave * 16;

  const _Float16* __restrict__ qlrb = qlr + (long)bh * SEQ * RANK;
  const _Float16* __restrict__ klrb = klr + (long)bh * SEQ * RANK;
  const _Float16* __restrict__ vtb  = vt  + (long)bh * DIM * SEQ;
  _Float16* __restrict__ pbase = &plds[wave][0][0];

  // Q A-tile (16x32 f16): lane row m = ll; K contiguous lh*16 .. lh*16+15.
  const _Float16* qr = qlrb + (long)(m0 + ll) * RANK + lh * 16;
  const v16h aQ = make16(*(const v8h*)qr, *(const v8h*)(qr + 8));

  v16h bOnes;
#pragma unroll
  for (int i = 0; i < 16; ++i) bOnes[i] = (_Float16)1.0f;

  v8f O[4] = {{}, {}, {}, {}};
  v8f Lacc = {};
  float mrow[8];
#pragma unroll
  for (int j = 0; j < 8; ++j) mrow[j] = NEG_BIG;

  const int ndiag = (m0 >> 6) << 6;   // start of the (single) masked chunk
  for (int nc = 0; nc < ndiag; nc += 64) {
    __builtin_prefetch(klrb + (long)(nc + 64 + ll) * RANK, 0, 3);
    __builtin_prefetch(vtb + (long)ll * SEQ + nc + 64, 0, 3);
    lsr_chunk<false>(nc, m0, lh, ll, klrb, vtb, aQ, bOnes, pbase, O, Lacc, mrow);
  }
  lsr_chunk<true>(ndiag, m0, lh, ll, klrb, vtb, aQ, bOnes, pbase, O, Lacc, mrow);

  // ---- epilogue: normalize by L (replicated across the 16-lane group).
  float* __restrict__ ob = out + (long)bh * SEQ * DIM;
#pragma unroll
  for (int j = 0; j < 8; ++j) {
    const int m = m0 + j + 8 * lh;
    const float inv = 1.0f / Lacc[j];
    float* orow = ob + (long)m * DIM + ll;
    orow[0]  = O[0][j] * inv;
    orow[16] = O[1][j] * inv;
    orow[32] = O[2][j] * inv;
    orow[48] = O[3][j] * inv;
  }
}

// ---------------------------------------------------------------------------
extern "C" void kernel_launch(void* const* d_in, const int* in_sizes, int n_in,
                              void* d_out, int out_size, void* d_ws, size_t ws_size,
                              hipStream_t stream) {
  (void)in_sizes; (void)n_in; (void)out_size; (void)ws_size;

  const float* q  = (const float*)d_in[0];
  const float* k  = (const float*)d_in[1];
  const float* v  = (const float*)d_in[2];
  const float* Wq = (const float*)d_in[3];
  const float* Wk = (const float*)d_in[4];
  const float* c1 = (const float*)d_in[5];
  const float* c2 = (const float*)d_in[6];
  float* out      = (float*)d_out;

  char* ws = (char*)d_ws;
  const size_t lr_bytes = (size_t)BH * SEQ * RANK * sizeof(_Float16);   // 4 MiB each
  _Float16* qlr = (_Float16*)(ws);
  _Float16* klr = (_Float16*)(ws + lr_bytes);
  _Float16* vt  = (_Float16*)(ws + 2 * lr_bytes);                       // 8 MiB

  lsr_proj_kernel<<<dim3((BH * SEQ) / 8), dim3(256), 0, stream>>>(
      q, k, Wq, Wk, c1, c2, qlr, klr);

  lsr_vtrans_kernel<<<dim3(SEQ / 64, BH), dim3(256), 0, stream>>>(v, vt);

  lsr_attn_kernel<<<dim3(SEQ / 64, BH), dim3(128), 0, stream>>>(qlr, klr, vt, out);
}
